// ECELoss_80865644249832
// MI455X (gfx1250) — matrix operand order repaired and checked by reading, not compiled, not run on hardware
//
#include <hip/hip_runtime.h>
#include <hip/hip_bf16.h>
#include <stdint.h>

typedef __attribute__((ext_vector_type(16))) _Float16 v16h;
typedef __attribute__((ext_vector_type(8)))  float    v8f;
typedef __attribute__((ext_vector_type(2)))  float    v2f;

#define ECE_NBINS 30
#define ECE_BLOCK 512
#define ECE_WAVES (ECE_BLOCK / 32)
#define ECE_GRID  512

union ABFrag { uint32_t w[8]; v16h v; };

// ---------------------------------------------------------------------------
// Kernel 1: zero the 96-float global accumulator (conf_rel[32], acc[32], cnt[32])
// ---------------------------------------------------------------------------
__global__ void ece_init_kernel(float* __restrict__ gacc) {
    int t = threadIdx.x;
    if (t < 96) gacc[t] = 0.0f;
}

// ---------------------------------------------------------------------------
// Kernel 2: main streaming pass. One row per lane per iteration; per-wave
// one-hot WMMA histogram (contention-free), block LDS reduce, global atomics.
// ---------------------------------------------------------------------------
__global__ __launch_bounds__(ECE_BLOCK) void ece_main_kernel(
        const float* __restrict__ sm,      // [n,10]
        const int*   __restrict__ labels,  // [n]
        float*       __restrict__ gacc,    // [96]
        int n) {
    __shared__ __align__(16) uint32_t      s_vals[ECE_WAVES][32]; // {acc_h16, conf_h16}
    __shared__ __align__(16) unsigned char s_bins[ECE_WAVES][32];
    __shared__ float s_hist[3][32];                                // stat x bin

    const int tid  = threadIdx.x;
    const int lane = tid & 31;
    const int wid  = tid >> 5;
    const int half = lane >> 4;          // 0: lanes 0-15, 1: lanes 16-31
    const uint32_t m = (uint32_t)(lane & 15);

    if (tid < 96) ((float*)s_hist)[tid] = 0.0f;
    __syncthreads();

    v8f acc0 = {};   // per-bin sums, bins 0..15
    v8f acc1 = {};   // per-bin sums, bins 16..31 (30,31 stay zero)

    const long long stride = (long long)gridDim.x * ECE_BLOCK;
    for (long long base = (long long)blockIdx.x * ECE_BLOCK; base < (long long)n;
         base += stride) {
        const long long row = base + tid;
        const bool valid = row < (long long)n;

        uint32_t packed = 0u;
        unsigned char bin_byte = 0xFF;
        if (valid) {
            const v2f* p2 = (const v2f*)(sm + row * 10);
            float v[10];
            v2f q0 = __builtin_nontemporal_load(p2 + 0);
            v2f q1 = __builtin_nontemporal_load(p2 + 1);
            v2f q2 = __builtin_nontemporal_load(p2 + 2);
            v2f q3 = __builtin_nontemporal_load(p2 + 3);
            v2f q4 = __builtin_nontemporal_load(p2 + 4);
            v[0]=q0.x; v[1]=q0.y; v[2]=q1.x; v[3]=q1.y; v[4]=q2.x;
            v[5]=q2.y; v[6]=q3.x; v[7]=q3.y; v[8]=q4.x; v[9]=q4.y;

            float best = v[0]; int arg = 0;
            #pragma unroll
            for (int i = 1; i < 10; i++) {
                if (v[i] > best) { best = v[i]; arg = i; }  // first-max wins ties
            }

            int b = (int)ceilf(best * 30.0f) - 1;           // (lower, upper] bins
            b = b < 0 ? 0 : (b > ECE_NBINS - 1 ? ECE_NBINS - 1 : b);
            // store confidence relative to bin center: |x| <= 1/60, f16-exactish
            float conf_rel = best - ((float)b + 0.5f) * (1.0f / 30.0f);
            _Float16 hcr = (_Float16)conf_rel;
            unsigned short crb = __builtin_bit_cast(unsigned short, hcr);
            int lab = labels[row];
            packed = (uint32_t)crb | ((arg == lab) ? 0x3C000000u : 0u); // acc f16 1.0
            bin_byte = (unsigned char)b;
        }

        s_vals[wid][lane] = packed;
        s_bins[wid][lane] = bin_byte;
        __syncthreads();

        // ---- Build one-hot A fragments (16-bit A 16x32 layout, ISA 7.12.2) ----
        // lanes 0-15 (M=lane):  VGPR j<4 -> K=2j,2j+1 ; VGPR 4+j -> K=16+2j,16+2j+1
        // lanes 16-31 (M=l-16): same with +8 K offset
        const uint64_t* bp = (const uint64_t*)(&s_bins[wid][0]);
        uint64_t g0 = bp[half];       // bins for K = 8*half .. 8*half+7
        uint64_t g1 = bp[2 + half];   // bins for K = 16+8*half .. +7
        ABFrag a0, a1, bb;
        const uint32_t m1 = m + 16u;
        #pragma unroll
        for (int j = 0; j < 4; j++) {
            uint32_t lo0 = (uint32_t)(g0 >> (16 * j))     & 0xFFu;
            uint32_t hi0 = (uint32_t)(g0 >> (16 * j + 8)) & 0xFFu;
            uint32_t lo1 = (uint32_t)(g1 >> (16 * j))     & 0xFFu;
            uint32_t hi1 = (uint32_t)(g1 >> (16 * j + 8)) & 0xFFu;
            a0.w[j]     = (lo0 == m  ? 0x3C00u : 0u) | (hi0 == m  ? 0x3C000000u : 0u);
            a0.w[4 + j] = (lo1 == m  ? 0x3C00u : 0u) | (hi1 == m  ? 0x3C000000u : 0u);
            a1.w[j]     = (lo0 == m1 ? 0x3C00u : 0u) | (hi0 == m1 ? 0x3C000000u : 0u);
            a1.w[4 + j] = (lo1 == m1 ? 0x3C00u : 0u) | (hi1 == m1 ? 0x3C000000u : 0u);
        }

        // ---- Build B fragment (32x16, N = lane%16, K pairs per VGPR) ----
        // col 0: conf_rel, col 1: correct, col 2: 1.0 (count), others: 0
        const int kb = half * 8;
        if (m < 2u) {
            const uint32_t sh = m * 16u;   // 0 -> conf (low half), 1 -> acc (high half)
            #pragma unroll
            for (int j = 0; j < 8; j++) {
                uint32_t v0 = s_vals[wid][kb + 2 * j + ((j >> 2) << 3)];
                uint32_t v1 = s_vals[wid][kb + 2 * j + ((j >> 2) << 3) + 1];
                bb.w[j] = ((v0 >> sh) & 0xFFFFu) | (((v1 >> sh) & 0xFFFFu) << 16);
            }
        } else if (m == 2u) {
            #pragma unroll
            for (int j = 0; j < 8; j++) bb.w[j] = 0x3C003C00u;  // (1.0h, 1.0h)
        } else {
            #pragma unroll
            for (int j = 0; j < 8; j++) bb.w[j] = 0u;
        }

        // EXEC is all-1s here (reconverged, uniform loop) as WMMA requires.
        acc0 = __builtin_amdgcn_wmma_f32_16x16x32_f16(
                   false, a0.v, false, bb.v, (short)0, acc0, false, false);
        acc1 = __builtin_amdgcn_wmma_f32_16x16x32_f16(
                   false, a1.v, false, bb.v, (short)0, acc1, false, false);
        __syncthreads();
    }

    // ---- Block reduction: D layout = lanes 0-15 N=lane (M=vgpr), 16-31 M=vgpr+8
    if (m < 3u) {
        #pragma unroll
        for (int vj = 0; vj < 8; vj++) {
            atomicAdd(&s_hist[m][8 * half + vj],      acc0[vj]);
            atomicAdd(&s_hist[m][16 + 8 * half + vj], acc1[vj]);
        }
    }
    __syncthreads();
    if (tid < 96) {
        float s = ((const float*)s_hist)[tid];
        atomicAdd(&gacc[tid], s);
    }
}

// ---------------------------------------------------------------------------
// Kernel 3: finalize ECE from the 96 accumulators.
// ---------------------------------------------------------------------------
__global__ void ece_final_kernel(const float* __restrict__ gacc,
                                 float* __restrict__ out, int n) {
    if (threadIdx.x == 0 && blockIdx.x == 0) {
        float ece = 0.0f;
        const float inv_n = 1.0f / (float)n;
        for (int b = 0; b < ECE_NBINS; b++) {
            float cnt = gacc[64 + b];
            if (cnt > 0.0f) {
                float avg_conf = ((float)b + 0.5f) * (1.0f / 30.0f) + gacc[b] / cnt;
                float acc_bin  = gacc[32 + b] / cnt;
                ece += fabsf(avg_conf - acc_bin) * cnt * inv_n;
            }
        }
        out[0] = ece;
    }
}

// ---------------------------------------------------------------------------
extern "C" void kernel_launch(void* const* d_in, const int* in_sizes, int n_in,
                              void* d_out, int out_size, void* d_ws, size_t ws_size,
                              hipStream_t stream) {
    const float* sm     = (const float*)d_in[0];
    const int*   labels = (const int*)d_in[1];
    float* gacc = (float*)d_ws;          // 96 floats scratch
    float* out  = (float*)d_out;         // 1 float
    const int n = in_sizes[1];           // labels count == N rows

    ece_init_kernel<<<1, 128, 0, stream>>>(gacc);
    ece_main_kernel<<<ECE_GRID, ECE_BLOCK, 0, stream>>>(sm, labels, gacc, n);
    ece_final_kernel<<<1, 32, 0, stream>>>(gacc, out, n);
}
